// DynamicSCMLearner_2267742732425
// MI455X (gfx1250) — compile-verified, matrix operand order checked by reading, not run on hardware
//
#include <hip/hip_runtime.h>
#include <hip/hip_bf16.h>
#include <math.h>

// Problem constants (match reference)
#define B_    64
#define HID_  1024
#define D_    64
#define VD_   64
#define SEMH_ 128
#define DVD_  4096   // D*VD
#define LDK   40     // LDS pitch in halves (80B): 16B-aligned frag loads, no bank conflicts

typedef __attribute__((ext_vector_type(16))) _Float16 v16h;
typedef __attribute__((ext_vector_type(8)))  _Float16 h8;
typedef __attribute__((ext_vector_type(2)))  _Float16 h2;
typedef __attribute__((ext_vector_type(8)))  float    v8f;
typedef __attribute__((ext_vector_type(4)))  float    f32x4;

__device__ __forceinline__ float gelu_exact(float x) {
  return 0.5f * x * (1.0f + erff(x * 0.70710678118654752440f));
}
__device__ __forceinline__ float sigmoid_(float x) {
  return 1.0f / (1.0f + expf(-x));
}

// v_wmma_f32_16x16x32_f16 A/B fragment from an LDS row (row-major, pitch LDK halves).
// Lane l<16 holds K = {0..7, 16..23}; lane>=16 holds K = {8..15, 24..31}.
// => two aligned 16-byte ds_load_b128 at half-offset 8*hi and 8*hi+16.
__device__ __forceinline__ v16h lds_frag(const _Float16* rowp, int hi) {
  const h8* q = (const h8*)(rowp + 8 * hi);
  h8 lo = q[0];
  h8 up = q[2];
  v16h a;
#pragma unroll
  for (int e = 0; e < 8; ++e) { a[e] = lo[e]; a[8 + e] = up[e]; }
  return a;
}

// ---------------------------------------------------------------------------
// Generic M=64 GEMM: C(64xN) = act(A(64xK) @ B(KxN) + bias), fp16 WMMA, fp32 acc.
// grid.x = N/128, block = 256 (8 waves). Double-buffered LDS; vectorized staging:
// A: 2x global_load_b128 + 1x ds_store_b128 per thread per tile
// B: 4x global_load_b128 + 8x ds_store_b32 (packed K-pairs) per thread per tile
// ---------------------------------------------------------------------------
template <bool GELU>
__global__ __launch_bounds__(256) void gemm64_wmma(
    const float* __restrict__ A, const float* __restrict__ Bm,
    const float* __restrict__ bias, float* __restrict__ C, int N, int K) {
  __shared__ __align__(16) _Float16 sA[2][64 * LDK];
  __shared__ __align__(16) _Float16 sB[2][128 * LDK];
  const int t = threadIdx.x;
  const int lane = t & 31, wv = t >> 5;
  const int l15 = lane & 15, hi = lane >> 4;
  const int n0 = blockIdx.x * 128;
  const int ar = t >> 2, ac = (t & 3) * 8;          // A: row, col-chunk of 8
  const int bn = (t & 15) * 8, bk = (t >> 4) * 2;   // B: n-chunk of 8, K pair
  v8f acc[4] = {};

  auto stage = [&](int buf, int kk) {
    const float* ap = A + (size_t)ar * K + kk + ac;
    f32x4 a0 = *(const f32x4*)(ap);
    f32x4 a1 = *(const f32x4*)(ap + 4);
    const float* b0p = Bm + (size_t)(kk + bk) * N + n0 + bn;
    const float* b1p = Bm + (size_t)(kk + bk + 1) * N + n0 + bn;
    f32x4 b00 = *(const f32x4*)(b0p);
    f32x4 b01 = *(const f32x4*)(b0p + 4);
    f32x4 b10 = *(const f32x4*)(b1p);
    f32x4 b11 = *(const f32x4*)(b1p + 4);
    h8 av;
#pragma unroll
    for (int j = 0; j < 4; ++j) { av[j] = (_Float16)a0[j]; av[4 + j] = (_Float16)a1[j]; }
    *(h8*)&sA[buf][ar * LDK + ac] = av;
#pragma unroll
    for (int j = 0; j < 8; ++j) {
      float x0 = (j < 4) ? b00[j] : b01[j - 4];
      float x1 = (j < 4) ? b10[j] : b11[j - 4];
      h2 hv; hv[0] = (_Float16)x0; hv[1] = (_Float16)x1;
      *(h2*)&sB[buf][(bn + j) * LDK + bk] = hv;     // 4B-aligned (bk even)
    }
  };

  stage(0, 0);
  __syncthreads();
  for (int kk = 0; kk < K; kk += 32) {
    const int cur = (kk >> 5) & 1;
    if (kk + 32 < K) stage(cur ^ 1, kk + 32);       // loads overlap compute below
    if (kk + 64 < K) {
      __builtin_prefetch(A + (size_t)ar * K + kk + 64 + ac, 0, 1);
      __builtin_prefetch(Bm + (size_t)(kk + 64 + bk) * N + n0 + bn, 0, 1);
    }
    const _Float16* cA = sA[cur];
    const _Float16* cB = sB[cur];
    v16h bf = lds_frag(cB + (wv * 16 + l15) * LDK, hi);
#pragma unroll
    for (int mt = 0; mt < 4; ++mt) {
      v16h af = lds_frag(cA + (mt * 16 + l15) * LDK, hi);
      acc[mt] = __builtin_amdgcn_wmma_f32_16x16x32_f16(
          false, af, false, bf, (short)0, acc[mt], false, false);
    }
    __syncthreads();                                // one barrier per K-step
  }
  const int col = n0 + wv * 16 + l15;
  const float bc = bias[col];
#pragma unroll
  for (int mt = 0; mt < 4; ++mt) {
#pragma unroll
    for (int r = 0; r < 8; ++r) {
      int row = mt * 16 + r + 8 * hi;               // C/D: VGPR r -> M=r (+8 for hi lanes)
      float v = acc[mt][r] + bc;
      if (GELU) v = gelu_exact(v);
      C[(size_t)row * N + col] = v;
    }
  }
}

// ---------------------------------------------------------------------------
// Group LayerNorm: 4096 groups of 64 (one wave per group), wave32 shfl reduce.
// ---------------------------------------------------------------------------
__global__ __launch_bounds__(256) void ln_groups(
    const float* __restrict__ X, const float* __restrict__ g,
    const float* __restrict__ b, float* __restrict__ Y) {
  int t = threadIdx.x, lane = t & 31, wv = t >> 5;
  int grp = blockIdx.x * 8 + wv;
  const float* xp = X + (size_t)grp * 64;
  float v0 = xp[lane], v1 = xp[lane + 32];
  float s = v0 + v1, q = v0 * v0 + v1 * v1;
#pragma unroll
  for (int m = 16; m >= 1; m >>= 1) {
    s += __shfl_xor(s, m, 32);
    q += __shfl_xor(q, m, 32);
  }
  float mean = s * (1.0f / 64.0f);
  float var = q * (1.0f / 64.0f) - mean * mean;
  float inv = rsqrtf(var + 1e-5f);
  float* yp = Y + (size_t)grp * 64;
  yp[lane]      = (v0 - mean) * inv * g[lane]      + b[lane];
  yp[lane + 32] = (v1 - mean) * inv * g[lane + 32] + b[lane + 32];
}

// ---------------------------------------------------------------------------
// Adjacency: sigmoid(5*W_masked), top-4 hard mask, M = W.^2, row |W| sums.
// ---------------------------------------------------------------------------
__global__ __launch_bounds__(64) void adjacency_k(
    const float* __restrict__ Wraw, float* __restrict__ adj_soft,
    float* __restrict__ adj_hard, float* __restrict__ Mbuf,
    float* __restrict__ rowAbs) {
  int i = threadIdx.x;
  float sab = 0.f;
  for (int j = 0; j < 64; ++j) {
    float ww = (j == i) ? 0.f : Wraw[i * 64 + j];
    sab += fabsf(ww);
    adj_soft[i * 64 + j] = sigmoid_(5.f * ww);
    Mbuf[i * 64 + j] = ww * ww;
    adj_hard[i * 64 + j] = 0.f;
  }
  rowAbs[i] = sab;
  int chosen[4] = {-1, -1, -1, -1};
#pragma unroll
  for (int p = 0; p < 4; ++p) {
    float bv = -1e30f;
    int bi = 0;
    for (int j = 0; j < 64; ++j) {
      bool skip = false;
#pragma unroll
      for (int q = 0; q < 4; ++q) skip = skip || (chosen[q] == j);
      if (skip) continue;
      float ww = (j == i) ? 0.f : Wraw[i * 64 + j];
      float s = sigmoid_(5.f * ww);
      if (s > bv) { bv = s; bi = j; }
    }
    chosen[p] = bi;
    adj_hard[i * 64 + bi] = (bv > 0.5f) ? 1.f : 0.f;
  }
}

// ---------------------------------------------------------------------------
// causal_loss = 1.0*(trace(expm(M)) - D) + 0.01*sum|W|
// trace(expm)-D = sum_{k>=1} tr(M^k)/k!  — 24-term Taylor in LDS.
// ---------------------------------------------------------------------------
__global__ __launch_bounds__(256) void expm_trace_k(
    const float* __restrict__ Mbuf, const float* __restrict__ rowAbs,
    float* __restrict__ loss) {
  __shared__ float sM[4096];
  __shared__ float sP[4096];
  __shared__ float sQ[4096];
  __shared__ float trAcc;
  int t = threadIdx.x;
  for (int idx = t; idx < 4096; idx += 256) {
    float m = Mbuf[idx];
    sM[idx] = m;
    sP[idx] = m;
  }
  if (t == 0) trAcc = 0.f;
  __syncthreads();
  float* cur = sP;
  float* nxt = sQ;
  for (int k = 1; k <= 24; ++k) {
    if (t == 0) {
      float tr = 0.f;
      for (int d = 0; d < 64; ++d) tr += cur[d * 64 + d];
      trAcc += tr;
    }
    float invk = 1.0f / (float)(k + 1);
    for (int idx = t; idx < 4096; idx += 256) {  // nxt = cur @ M / (k+1)
      int r = idx >> 6, c = idx & 63;
      float s = 0.f;
      for (int j = 0; j < 64; ++j) s = fmaf(cur[r * 64 + j], sM[j * 64 + c], s);
      nxt[idx] = s * invk;
    }
    __syncthreads();
    float* tmp = cur; cur = nxt; nxt = tmp;
  }
  if (t == 0) {
    float sp = 0.f;
    for (int d = 0; d < 64; ++d) sp += rowAbs[d];
    loss[0] = 1.0f * trAcc + 0.01f * sp;
  }
}

// ---------------------------------------------------------------------------
// SEM layer 1: per variable i, H[i] = gelu((cv * adj_row) @ W1_i + b1_i)
// adj mask folded into the f32->f16 conversion during A staging.
// grid=64 (per i), block=256. Double-buffered, vectorized staging.
// ---------------------------------------------------------------------------
__global__ __launch_bounds__(256) void sem1_wmma(
    const float* __restrict__ cv, const float* __restrict__ adj,
    const float* __restrict__ W1, const float* __restrict__ b1,
    float* __restrict__ H) {
  __shared__ __align__(16) _Float16 sA[2][64 * LDK];
  __shared__ __align__(16) _Float16 sB[2][128 * LDK];
  __shared__ float sAdj[64];
  const int i = blockIdx.x;
  const int t = threadIdx.x;
  const int lane = t & 31, wv = t >> 5, l15 = lane & 15, hi = lane >> 4;
  const int ar = t >> 2, ac = (t & 3) * 8;
  const int bn = (t & 15) * 8, bk = (t >> 4) * 2;
  if (t < 64) sAdj[t] = adj[i * 64 + t];
  __syncthreads();
  const float* Bm = W1 + (size_t)i * DVD_ * SEMH_;
  v8f acc[4] = {};

  auto stage = [&](int buf, int kk) {
    const float* ap = cv + (size_t)ar * DVD_ + kk + ac;
    f32x4 a0 = *(const f32x4*)(ap);
    f32x4 a1 = *(const f32x4*)(ap + 4);
    const float sc = sAdj[(kk + ac) >> 6];          // 8-chunk never crosses a VD group
    const float* b0p = Bm + (size_t)(kk + bk) * SEMH_ + bn;
    const float* b1p = Bm + (size_t)(kk + bk + 1) * SEMH_ + bn;
    f32x4 b00 = *(const f32x4*)(b0p);
    f32x4 b01 = *(const f32x4*)(b0p + 4);
    f32x4 b10 = *(const f32x4*)(b1p);
    f32x4 b11 = *(const f32x4*)(b1p + 4);
    h8 av;
#pragma unroll
    for (int j = 0; j < 4; ++j) {
      av[j]     = (_Float16)(a0[j] * sc);
      av[4 + j] = (_Float16)(a1[j] * sc);
    }
    *(h8*)&sA[buf][ar * LDK + ac] = av;
#pragma unroll
    for (int j = 0; j < 8; ++j) {
      float x0 = (j < 4) ? b00[j] : b01[j - 4];
      float x1 = (j < 4) ? b10[j] : b11[j - 4];
      h2 hv; hv[0] = (_Float16)x0; hv[1] = (_Float16)x1;
      *(h2*)&sB[buf][(bn + j) * LDK + bk] = hv;
    }
  };

  stage(0, 0);
  __syncthreads();
  for (int kk = 0; kk < DVD_; kk += 32) {
    const int cur = (kk >> 5) & 1;
    if (kk + 32 < DVD_) stage(cur ^ 1, kk + 32);
    if (kk + 64 < DVD_)
      __builtin_prefetch(Bm + (size_t)(kk + 64 + bk) * SEMH_ + bn, 0, 1);
    const _Float16* cA = sA[cur];
    const _Float16* cB = sB[cur];
    v16h bf = lds_frag(cB + (wv * 16 + l15) * LDK, hi);
#pragma unroll
    for (int mt = 0; mt < 4; ++mt) {
      v16h af = lds_frag(cA + (mt * 16 + l15) * LDK, hi);
      acc[mt] = __builtin_amdgcn_wmma_f32_16x16x32_f16(
          false, af, false, bf, (short)0, acc[mt], false, false);
    }
    __syncthreads();
  }
  const int col = wv * 16 + l15;
  const float bc = b1[i * SEMH_ + col];
#pragma unroll
  for (int mt = 0; mt < 4; ++mt)
#pragma unroll
    for (int r = 0; r < 8; ++r) {
      int row = mt * 16 + r + 8 * hi;
      H[(size_t)i * (64 * SEMH_) + (size_t)row * SEMH_ + col] =
          gelu_exact(acc[mt][r] + bc);
    }
}

// ---------------------------------------------------------------------------
// SEM layer 2 + bias + per-(b,i) LayerNorm -> updated (written into d_out).
// grid=64 (per i), block=128 (4 waves); 64x64 output, K=128 WMMA, double-buffered.
// ---------------------------------------------------------------------------
__global__ __launch_bounds__(128) void sem2_wmma(
    const float* __restrict__ H, const float* __restrict__ W2,
    const float* __restrict__ b2, const float* __restrict__ slng,
    const float* __restrict__ slnb, float* __restrict__ updated) {
  __shared__ __align__(16) _Float16 sA[2][64 * LDK];
  __shared__ __align__(16) _Float16 sB[2][64 * LDK];
  __shared__ float sO[64 * 65];                     // pitch 65 -> conflict-free row scans
  __shared__ float sMean[64], sInv[64];
  const int i = blockIdx.x;
  const int t = threadIdx.x;
  const int lane = t & 31, wv = t >> 5, l15 = lane & 15, hi = lane >> 4;
  const int ar = t >> 1, ac = (t & 1) * 16;         // A: 16 floats per thread
  const int bn = (t & 7) * 8, bk = (t >> 3) * 2;    // B: n-chunk of 8, K pair
  const float* Am = H + (size_t)i * (64 * SEMH_);
  const float* Bm = W2 + (size_t)i * (SEMH_ * VD_);
  v8f acc[4] = {};

  auto stage = [&](int buf, int kk) {
    const float* ap = Am + (size_t)ar * SEMH_ + kk + ac;
    f32x4 a0 = *(const f32x4*)(ap);
    f32x4 a1 = *(const f32x4*)(ap + 4);
    f32x4 a2 = *(const f32x4*)(ap + 8);
    f32x4 a3 = *(const f32x4*)(ap + 12);
    const float* b0p = Bm + (size_t)(kk + bk) * VD_ + bn;
    const float* b1p = Bm + (size_t)(kk + bk + 1) * VD_ + bn;
    f32x4 b00 = *(const f32x4*)(b0p);
    f32x4 b01 = *(const f32x4*)(b0p + 4);
    f32x4 b10 = *(const f32x4*)(b1p);
    f32x4 b11 = *(const f32x4*)(b1p + 4);
    h8 av0, av1;
#pragma unroll
    for (int j = 0; j < 4; ++j) {
      av0[j] = (_Float16)a0[j]; av0[4 + j] = (_Float16)a1[j];
      av1[j] = (_Float16)a2[j]; av1[4 + j] = (_Float16)a3[j];
    }
    *(h8*)&sA[buf][ar * LDK + ac] = av0;
    *(h8*)&sA[buf][ar * LDK + ac + 8] = av1;
#pragma unroll
    for (int j = 0; j < 8; ++j) {
      float x0 = (j < 4) ? b00[j] : b01[j - 4];
      float x1 = (j < 4) ? b10[j] : b11[j - 4];
      h2 hv; hv[0] = (_Float16)x0; hv[1] = (_Float16)x1;
      *(h2*)&sB[buf][(bn + j) * LDK + bk] = hv;
    }
  };

  stage(0, 0);
  __syncthreads();
  for (int kk = 0; kk < SEMH_; kk += 32) {
    const int cur = (kk >> 5) & 1;
    if (kk + 32 < SEMH_) stage(cur ^ 1, kk + 32);
    const _Float16* cA = sA[cur];
    const _Float16* cB = sB[cur];
    v16h bf = lds_frag(cB + (wv * 16 + l15) * LDK, hi);
#pragma unroll
    for (int mt = 0; mt < 4; ++mt) {
      v16h af = lds_frag(cA + (mt * 16 + l15) * LDK, hi);
      acc[mt] = __builtin_amdgcn_wmma_f32_16x16x32_f16(
          false, af, false, bf, (short)0, acc[mt], false, false);
    }
    __syncthreads();
  }
  const int col = wv * 16 + l15;                    // v in 0..63
  const float bc = b2[i * VD_ + col];
#pragma unroll
  for (int mt = 0; mt < 4; ++mt)
#pragma unroll
    for (int r = 0; r < 8; ++r) {
      int row = mt * 16 + r + 8 * hi;
      sO[row * 65 + col] = acc[mt][r] + bc;
    }
  __syncthreads();
  if (t < 64) {                                     // LN stats per batch row
    float s = 0.f, q = 0.f;
    for (int v = 0; v < 64; ++v) {
      float x = sO[t * 65 + v];
      s += x; q += x * x;
    }
    float mean = s * (1.f / 64.f);
    sMean[t] = mean;
    sInv[t] = rsqrtf(q * (1.f / 64.f) - mean * mean + 1e-5f);
  }
  __syncthreads();
  for (int idx = t; idx < 4096; idx += 128) {
    int b = idx >> 6, v = idx & 63;
    float x = (sO[b * 65 + v] - sMean[b]) * sInv[b] * slng[i * VD_ + v] + slnb[i * VD_ + v];
    updated[(size_t)b * DVD_ + i * VD_ + v] = x;
  }
}

// ---------------------------------------------------------------------------
extern "C" void kernel_launch(void* const* d_in, const int* in_sizes, int n_in,
                              void* d_out, int out_size, void* d_ws, size_t ws_size,
                              hipStream_t stream) {
  (void)in_sizes; (void)n_in; (void)out_size; (void)ws_size;
  const float* pooled = (const float*)d_in[0];
  const float* p1w    = (const float*)d_in[1];
  const float* p1b    = (const float*)d_in[2];
  const float* p2w    = (const float*)d_in[3];
  const float* p2b    = (const float*)d_in[4];
  const float* plng   = (const float*)d_in[5];
  const float* plnb   = (const float*)d_in[6];
  const float* Wraw   = (const float*)d_in[7];
  const float* semW1  = (const float*)d_in[8];
  const float* semb1  = (const float*)d_in[9];
  const float* semW2  = (const float*)d_in[10];
  const float* semb2  = (const float*)d_in[11];
  const float* slng   = (const float*)d_in[12];
  const float* slnb   = (const float*)d_in[13];
  const float* outw   = (const float*)d_in[14];
  const float* outb   = (const float*)d_in[15];

  float* out = (float*)d_out;                // outputs concatenated flat
  float* causal_repr = out;                  // 64*1024   = 65536
  float* updated     = out + 65536;          // 64*64*64  = 262144
  float* adj_soft    = out + 65536 + 262144; // 64*64     = 4096
  float* adj_hard    = adj_soft + 4096;      // 64*64     = 4096
  float* loss        = adj_hard + 4096;      // 1

  float* ws     = (float*)d_ws;
  float* t1     = ws;                        // 65536
  float* xbuf   = t1 + 65536;                // 262144
  float* cv     = xbuf + 262144;             // 262144
  float* Mbuf   = cv + 262144;               // 4096
  float* rowAbs = Mbuf + 4096;               // 64
  float* Hbuf   = rowAbs + 64;               // 64*64*128 = 524288

  gemm64_wmma<true ><<<dim3(8),  dim3(256), 0, stream>>>(pooled, p1w, p1b, t1, HID_, HID_);
  gemm64_wmma<false><<<dim3(32), dim3(256), 0, stream>>>(t1, p2w, p2b, xbuf, DVD_, HID_);
  ln_groups        <<<dim3(512), dim3(256), 0, stream>>>(xbuf, plng, plnb, cv);
  adjacency_k      <<<dim3(1),   dim3(64),  0, stream>>>(Wraw, adj_soft, adj_hard, Mbuf, rowAbs);
  expm_trace_k     <<<dim3(1),   dim3(256), 0, stream>>>(Mbuf, rowAbs, loss);
  sem1_wmma        <<<dim3(64),  dim3(256), 0, stream>>>(cv, adj_soft, semW1, semb1, Hbuf);
  sem2_wmma        <<<dim3(64),  dim3(128), 0, stream>>>(Hbuf, semW2, semb2, slng, slnb, updated);
  gemm64_wmma<false><<<dim3(8),  dim3(256), 0, stream>>>(updated, outw, outb, causal_repr, HID_, DVD_);
}
